// RowSelfAttention_12506944766491
// MI455X (gfx1250) — matrix operand order repaired and checked by reading, not compiled, not run on hardware
//
#include <hip/hip_runtime.h>

#define EMBED 768
#define HEADS 12
#define DKH   64
#define RR    128
#define CC    256
#define MTOT  (RR * CC)          // 32768 flattened (r,i) rows
#define NME   ((size_t)MTOT * EMBED)
#define NWE   ((size_t)EMBED * EMBED)

typedef __bf16 bf16;
typedef __attribute__((ext_vector_type(16))) __bf16 v16bf;
typedef __attribute__((ext_vector_type(8)))  __bf16 v8bf;
typedef __attribute__((ext_vector_type(8)))  float  v8f;
typedef __attribute__((ext_vector_type(4)))  int    i32x4;

// ---------------------------------------------------------------------------
// Async global->LDS copy (CDNA5 GLOBAL_LOAD_ASYNC_TO_LDS_B128, ASYNCcnt).
// Guarded: falls back to a synchronous 16B copy with identical call sites.
// ---------------------------------------------------------------------------
#if defined(__has_builtin)
#if __has_builtin(__builtin_amdgcn_global_load_async_to_lds_b128) && \
    __has_builtin(__builtin_amdgcn_s_wait_asynccnt)
#define HAVE_ASYNC_LDS 1
#endif
#endif
#ifndef HAVE_ASYNC_LDS
#define HAVE_ASYNC_LDS 0
#endif

#if HAVE_ASYNC_LDS
static __device__ __forceinline__ void async_cp16(const bf16* g, bf16* l) {
  // builtin prototype takes a NON-const global pointer; C-style cast drops
  // const and performs the addrspace cast.
  __builtin_amdgcn_global_load_async_to_lds_b128(
      (__attribute__((address_space(1))) i32x4*)g,
      (__attribute__((address_space(3))) i32x4*)l, 0, 0);
}
#define WAIT_ASYNC(n) __builtin_amdgcn_s_wait_asynccnt(n)
#else
static __device__ __forceinline__ void async_cp16(const bf16* g, bf16* l) {
  *(i32x4*)l = *(const i32x4*)g;
}
#define WAIT_ASYNC(n) ((void)0)
#endif

// ---------------------------------------------------------------------------
// WMMA fragment loaders (layouts per ISA §7.12.2, bf16 16x16x32).
// A (16x32): lane 0-15 -> M=lane, K={0..7,16..23}; lane16-31 -> K={8..15,24..31}
// B (32x16), LDS stored [n][k]: lane0-15 -> N=lane,K={0..7,16..23}; lane16-31 -> K={8..15,24..31}
// ---------------------------------------------------------------------------
static __device__ __forceinline__ v16bf load_frag_a(const bf16* row, int kbase) {
  v8bf lo = *(const v8bf*)(row + kbase);
  v8bf hi = *(const v8bf*)(row + kbase + 16);
  return __builtin_shufflevector(lo, hi, 0,1,2,3,4,5,6,7,8,9,10,11,12,13,14,15);
}
static __device__ __forceinline__ v16bf load_frag_b(const bf16* nrow, int kbase) {
  v8bf lo = *(const v8bf*)(nrow + kbase);
  v8bf hi = *(const v8bf*)(nrow + kbase + 16);
  return __builtin_shufflevector(lo, hi, 0,1,2,3,4,5,6,7,8,9,10,11,12,13,14,15);
}
static __device__ __forceinline__ v8f wmma_bf16(v16bf a, v16bf b, v8f c) {
  return __builtin_amdgcn_wmma_f32_16x16x32_bf16(false, a, false, b, (short)0, c,
                                                 false, false);
}

// ---------------------------------------------------------------------------
// fp32 -> bf16 conversion pass (vectorized, 8 elems/thread).
// ---------------------------------------------------------------------------
__global__ __launch_bounds__(256)
void cvt_bf16_kernel(const float* __restrict__ in, bf16* __restrict__ out, int n) {
  const int i = (blockIdx.x * 256 + threadIdx.x) * 8;
  if (i + 8 <= n) {
    float4 a = *(const float4*)(in + i);
    float4 b = *(const float4*)(in + i + 4);
    v8bf o;
    o[0] = (bf16)a.x; o[1] = (bf16)a.y; o[2] = (bf16)a.z; o[3] = (bf16)a.w;
    o[4] = (bf16)b.x; o[5] = (bf16)b.y; o[6] = (bf16)b.z; o[7] = (bf16)b.w;
    *(v8bf*)(out + i) = o;
  }
}

// ---------------------------------------------------------------------------
// GEMM (all bf16 in): out[m][n] = (sum_k A[m][k]*W[n][k] + bias[n]) * scale
// 128x128 block tile, K chunk 32, 8 waves x (32x64), async double-buffered LDS.
// ---------------------------------------------------------------------------
template <bool OUT_BF16>
__global__ __launch_bounds__(256)
void gemm_bb_kernel(const bf16* __restrict__ A, const bf16* __restrict__ W,
                    const float* __restrict__ bias, void* __restrict__ outp,
                    int M, int N, int K, float scale) {
  __shared__ __align__(16) bf16 sA[2][128 * 40];  // [m][k]
  __shared__ __align__(16) bf16 sB[2][128 * 40];  // [n][k]

  const int tid  = threadIdx.x;
  const int lane = tid & 31;
  const int wave = tid >> 5;
  const int bm = blockIdx.y * 128;
  const int bn = blockIdx.x * 128;
  const int wm = (wave & 3) * 32;
  const int wn = (wave >> 2) * 64;

  const int lrow = tid >> 1;
  const int lcol = (tid & 1) * 16;

  auto issue = [&](int k0, int buf) {
    const bf16* ga = A + (size_t)(bm + lrow) * K + k0 + lcol;
    bf16* la = &sA[buf][lrow * 40 + lcol];
    async_cp16(ga, la);
    async_cp16(ga + 8, la + 8);
    const bf16* gb = W + (size_t)(bn + lrow) * K + k0 + lcol;
    bf16* lb = &sB[buf][lrow * 40 + lcol];
    async_cp16(gb, lb);
    async_cp16(gb + 8, lb + 8);
  };

  v8f acc[2][4];
#pragma unroll
  for (int mt = 0; mt < 2; mt++)
#pragma unroll
    for (int nt = 0; nt < 4; nt++)
#pragma unroll
      for (int v = 0; v < 8; v++) acc[mt][nt][v] = 0.0f;

  const int NK = K / 32;
  issue(0, 0);
  for (int kc = 0; kc < NK; kc++) {
    const int cur = kc & 1;
    if (kc + 1 < NK) {
      issue((kc + 1) * 32, cur ^ 1);
      WAIT_ASYNC(4);       // 4 oldest (current tile) complete; in-order retire
    } else {
      WAIT_ASYNC(0);
    }
    __syncthreads();

    const int arow  = wm + (lane & 15);
    const int abase = (lane >> 4) * 8;
    const int bcol  = wn + (lane & 15);
    const int bbase = (lane >> 4) * 8;
    v16bf af[2], bfr[4];
#pragma unroll
    for (int mt = 0; mt < 2; mt++)
      af[mt] = load_frag_a(&sA[cur][(arow + mt * 16) * 40], abase);
#pragma unroll
    for (int nt = 0; nt < 4; nt++)
      bfr[nt] = load_frag_b(&sB[cur][(bcol + nt * 16) * 40], bbase);
#pragma unroll
    for (int mt = 0; mt < 2; mt++)
#pragma unroll
      for (int nt = 0; nt < 4; nt++)
        acc[mt][nt] = wmma_bf16(af[mt], bfr[nt], acc[mt][nt]);
    __syncthreads();  // buffer (cur) free for re-issue next iteration
  }

  const int orow0 = bm + wm + ((lane >> 4) << 3);
  const int ocol0 = bn + wn + (lane & 15);
#pragma unroll
  for (int mt = 0; mt < 2; mt++) {
#pragma unroll
    for (int nt = 0; nt < 4; nt++) {
      const int col = ocol0 + nt * 16;
      const float bv = bias[col];
#pragma unroll
      for (int v = 0; v < 8; v++) {
        const int row = orow0 + mt * 16 + v;
        const float rv = (acc[mt][nt][v] + bv) * scale;
        if constexpr (OUT_BF16)
          ((bf16*)outp)[(size_t)row * N + col] = (bf16)rv;
        else
          ((float*)outp)[(size_t)row * N + col] = rv;
      }
    }
  }
}

// ---------------------------------------------------------------------------
// Scores: S[h][i][j] = sum_{r,d} Q[r,i,h,d]*K[r,j,h,d]. 128x128 tile per head,
// r-loop with async double-buffered Q/K tiles (128x64 bf16 each).
// ---------------------------------------------------------------------------
__global__ __launch_bounds__(256)
void scores_kernel(const bf16* __restrict__ Qb, const bf16* __restrict__ Kb,
                   float* __restrict__ S) {
  __shared__ __align__(16) bf16 sQ[2][128 * 72];
  __shared__ __align__(16) bf16 sK[2][128 * 72];

  const int tid  = threadIdx.x;
  const int lane = tid & 31;
  const int wave = tid >> 5;
  const int bi = blockIdx.x * 128;
  const int bj = blockIdx.y * 128;
  const int h  = blockIdx.z;
  const int wm = (wave & 3) * 32;
  const int wn = (wave >> 2) * 64;

  const int lrow = tid >> 1;
  const int lcol = (tid & 1) * 32;

  auto issue = [&](int r, int buf) {
    const bf16* qs = Qb + ((size_t)((r * CC + bi + lrow) * HEADS + h)) * DKH + lcol;
    bf16* qd = &sQ[buf][lrow * 72 + lcol];
    async_cp16(qs, qd);           async_cp16(qs + 8, qd + 8);
    async_cp16(qs + 16, qd + 16); async_cp16(qs + 24, qd + 24);
    const bf16* ks = Kb + ((size_t)((r * CC + bj + lrow) * HEADS + h)) * DKH + lcol;
    bf16* kd = &sK[buf][lrow * 72 + lcol];
    async_cp16(ks, kd);           async_cp16(ks + 8, kd + 8);
    async_cp16(ks + 16, kd + 16); async_cp16(ks + 24, kd + 24);
  };

  v8f acc[2][4];
#pragma unroll
  for (int mt = 0; mt < 2; mt++)
#pragma unroll
    for (int nt = 0; nt < 4; nt++)
#pragma unroll
      for (int v = 0; v < 8; v++) acc[mt][nt][v] = 0.0f;

  issue(0, 0);
  for (int r = 0; r < RR; r++) {
    const int cur = r & 1;
    if (r + 1 < RR) {
      issue(r + 1, cur ^ 1);
      WAIT_ASYNC(8);
    } else {
      WAIT_ASYNC(0);
    }
    __syncthreads();

#pragma unroll
    for (int kc = 0; kc < 2; kc++) {
      const int kb = kc * 32;
      const int arow  = wm + (lane & 15);
      const int abase = kb + (lane >> 4) * 8;
      const int bcol  = wn + (lane & 15);
      const int bbase = kb + (lane >> 4) * 8;
      v16bf af[2], bfr[4];
#pragma unroll
      for (int mt = 0; mt < 2; mt++)
        af[mt] = load_frag_a(&sQ[cur][(arow + mt * 16) * 72], abase);
#pragma unroll
      for (int nt = 0; nt < 4; nt++)
        bfr[nt] = load_frag_b(&sK[cur][(bcol + nt * 16) * 72], bbase);
#pragma unroll
      for (int mt = 0; mt < 2; mt++)
#pragma unroll
        for (int nt = 0; nt < 4; nt++)
          acc[mt][nt] = wmma_bf16(af[mt], bfr[nt], acc[mt][nt]);
    }
    __syncthreads();
  }

  const int orow0 = bi + wm + ((lane >> 4) << 3);
  const int ocol0 = bj + wn + (lane & 15);
  float* Sh = S + (size_t)h * CC * CC;
#pragma unroll
  for (int mt = 0; mt < 2; mt++)
#pragma unroll
    for (int nt = 0; nt < 4; nt++)
#pragma unroll
      for (int v = 0; v < 8; v++)
        Sh[(size_t)(orow0 + mt * 16 + v) * CC + ocol0 + nt * 16] = acc[mt][nt][v];
}

// ---------------------------------------------------------------------------
// Softmax over j per (h,i). Writes fp32 probs (output #2) + bf16 copy for the
// context GEMM's async path.
// ---------------------------------------------------------------------------
__global__ __launch_bounds__(256)
void softmax_kernel(const float* __restrict__ S, float* __restrict__ P,
                    bf16* __restrict__ Pb) {
  __shared__ float red[256];
  const int row = blockIdx.x;  // h*256 + i
  const int j   = threadIdx.x;
  const float s = S[(size_t)row * CC + j];
  red[j] = s;
  __syncthreads();
  for (int st = 128; st > 0; st >>= 1) {
    if (j < st) red[j] = fmaxf(red[j], red[j + st]);
    __syncthreads();
  }
  const float m = red[0];
  __syncthreads();
  const float e = __expf(s - m);
  red[j] = e;
  __syncthreads();
  for (int st = 128; st > 0; st >>= 1) {
    if (j < st) red[j] += red[j + st];
    __syncthreads();
  }
  const float p = e / red[0];
  P[(size_t)row * CC + j]  = p;
  Pb[(size_t)row * CC + j] = (bf16)p;
}

// ---------------------------------------------------------------------------
// Context: C[r,i,h,d] = sum_j P[h,i,j]*V[r,j,h,d]. Block=(r,h): 256x64 output.
// V transposed into LDS once; bf16 P chunks async double-buffered.
// ---------------------------------------------------------------------------
__global__ __launch_bounds__(256)
void context_kernel(const bf16* __restrict__ Pb, const bf16* __restrict__ Vb,
                    bf16* __restrict__ Cb) {
  __shared__ __align__(16) bf16 sP[2][256 * 40];  // [i][j-chunk]
  __shared__ __align__(16) bf16 sVT[64 * 264];    // [d][j]

  const int tid  = threadIdx.x;
  const int lane = tid & 31;
  const int wave = tid >> 5;
  const int r = blockIdx.x;
  const int h = blockIdx.y;
  const int wm = (wave & 3) * 64;
  const int wn = (wave >> 2) * 32;

  const bf16* Ph = Pb + (size_t)h * CC * CC;
  auto issueP = [&](int jc, int buf) {
    const bf16* src = Ph + (size_t)tid * CC + jc * 32;
    bf16* dst = &sP[buf][tid * 40];
    async_cp16(src, dst);           async_cp16(src + 8, dst + 8);
    async_cp16(src + 16, dst + 16); async_cp16(src + 24, dst + 24);
  };

  issueP(0, 0);

  // V (256x64) transposed: thread j scatters its row into sVT[d][j].
  {
    const bf16* vsrc = Vb + ((size_t)((r * CC + tid) * HEADS + h)) * DKH;
#pragma unroll
    for (int d = 0; d < DKH; d++) sVT[d * 264 + tid] = vsrc[d];
  }

  v8f acc[4][2];
#pragma unroll
  for (int mt = 0; mt < 4; mt++)
#pragma unroll
    for (int nt = 0; nt < 2; nt++)
#pragma unroll
      for (int v = 0; v < 8; v++) acc[mt][nt][v] = 0.0f;

  for (int jc = 0; jc < 8; jc++) {
    const int cur = jc & 1;
    if (jc + 1 < 8) {
      issueP(jc + 1, cur ^ 1);
      WAIT_ASYNC(4);
    } else {
      WAIT_ASYNC(0);
    }
    __syncthreads();

    const int arow  = wm + (lane & 15);
    const int abase = (lane >> 4) * 8;
    const int bcol  = wn + (lane & 15);
    const int bbase = jc * 32 + (lane >> 4) * 8;
    v16bf af[4], bfr[2];
#pragma unroll
    for (int mt = 0; mt < 4; mt++)
      af[mt] = load_frag_a(&sP[cur][(arow + mt * 16) * 40], abase);
#pragma unroll
    for (int nt = 0; nt < 2; nt++)
      bfr[nt] = load_frag_b(&sVT[(bcol + nt * 16) * 264], bbase);
#pragma unroll
    for (int mt = 0; mt < 4; mt++)
#pragma unroll
      for (int nt = 0; nt < 2; nt++)
        acc[mt][nt] = wmma_bf16(af[mt], bfr[nt], acc[mt][nt]);
    __syncthreads();
  }

  const int orow0 = wm + ((lane >> 4) << 3);
  const int ocol0 = wn + (lane & 15);
#pragma unroll
  for (int mt = 0; mt < 4; mt++)
#pragma unroll
    for (int nt = 0; nt < 2; nt++)
#pragma unroll
      for (int v = 0; v < 8; v++) {
        const int row = orow0 + mt * 16 + v;  // i
        const int col = ocol0 + nt * 16;      // d
        Cb[((size_t)(r * CC + row) * HEADS + h) * DKH + col] = (bf16)acc[mt][nt][v];
      }
}

// ---------------------------------------------------------------------------
extern "C" void kernel_launch(void* const* d_in, const int* in_sizes, int n_in,
                              void* d_out, int out_size, void* d_ws, size_t ws_size,
                              hipStream_t stream) {
  const float* x  = (const float*)d_in[0];
  const float* Wq = (const float*)d_in[1];
  const float* bq = (const float*)d_in[2];
  const float* Wk = (const float*)d_in[3];
  const float* bk = (const float*)d_in[4];
  const float* Wv = (const float*)d_in[5];
  const float* bv = (const float*)d_in[6];
  const float* Wo = (const float*)d_in[7];
  const float* bo = (const float*)d_in[8];

  float* out   = (float*)d_out;               // (128,256,1,768) fp32
  float* probs = out + NME;                   // (12,1,256,256) fp32

  // workspace layout (bf16 unless noted):
  bf16* xb  = (bf16*)d_ws;       // x in bf16          (25.2M)
  bf16* Qb  = xb  + NME;
  bf16* Kb  = Qb  + NME;
  bf16* Vb  = Kb  + NME;
  bf16* Wqb = Vb  + NME;         // weights in bf16    (4 x 589824)
  bf16* Wkb = Wqb + NWE;
  bf16* Wvb = Wkb + NWE;
  bf16* Wob = Wvb + NWE;
  float* Sc = (float*)(Wob + NWE);            // scores fp32 (786432)
  bf16* Pb  = (bf16*)(Sc + (size_t)HEADS * CC * CC);  // probs bf16
  bf16* Cb  = xb;                // context reuses x buffer (x dead after QKV)

  const float qscale = 0.011048543456039806f; // Dk^-0.5 / sqrt(R)

  // 1) convert x and weights to bf16 once
  cvt_bf16_kernel<<<(int)(NME / 2048), 256, 0, stream>>>(x, xb, (int)NME);
  cvt_bf16_kernel<<<(int)(NWE / 2048), 256, 0, stream>>>(Wq, Wqb, (int)NWE);
  cvt_bf16_kernel<<<(int)(NWE / 2048), 256, 0, stream>>>(Wk, Wkb, (int)NWE);
  cvt_bf16_kernel<<<(int)(NWE / 2048), 256, 0, stream>>>(Wv, Wvb, (int)NWE);
  cvt_bf16_kernel<<<(int)(NWE / 2048), 256, 0, stream>>>(Wo, Wob, (int)NWE);

  // 2) QKV projections
  dim3 gP(EMBED / 128, MTOT / 128);  // (6, 256)
  gemm_bb_kernel<true><<<gP, 256, 0, stream>>>(xb, Wqb, bq, Qb, MTOT, EMBED, EMBED, qscale);
  gemm_bb_kernel<true><<<gP, 256, 0, stream>>>(xb, Wkb, bk, Kb, MTOT, EMBED, EMBED, 1.0f);
  gemm_bb_kernel<true><<<gP, 256, 0, stream>>>(xb, Wvb, bv, Vb, MTOT, EMBED, EMBED, 1.0f);

  // 3) scores + softmax
  dim3 gS(CC / 128, CC / 128, HEADS);  // (2, 2, 12)
  scores_kernel<<<gS, 256, 0, stream>>>(Qb, Kb, Sc);
  softmax_kernel<<<HEADS * CC, 256, 0, stream>>>(Sc, probs, Pb);

  // 4) context + output projection
  dim3 gC(RR, HEADS);  // (128, 12)
  context_kernel<<<gC, 256, 0, stream>>>(Pb, Vb, Cb);
  gemm_bb_kernel<false><<<gP, 256, 0, stream>>>(Cb, Wob, bo, out, MTOT, EMBED, EMBED, 1.0f);
}
